// CausalTradeGNN_292057776305
// MI455X (gfx1250) — matrix operand order repaired and checked by reading, not compiled, not run on hardware
//
#include <hip/hip_runtime.h>
#include <math.h>

// ---------------------------------------------------------------------------
// CausalTradeGNN forward for MI455X (gfx1250, wave32, WMMA).
// All dense layers via v_wmma_f32_16x16x32_f16 (f16 in, f32 acc).
// All WMMA operands pre-converted to zero-padded f16 so fragment loads are
// contiguous 16-byte b128 loads (no guards, no per-element cvt).
// Per-edge MLPs fully fused per 16-edge tile in LDS (no HBM staging).
// Irregular graph ops (segment softmax / scatter) use f32 global atomics.
// ---------------------------------------------------------------------------

#define N_NODES 10000
#define N_EDGES 200000
#define HIDC    64

typedef __attribute__((ext_vector_type(16))) _Float16 v16h;
typedef __attribute__((ext_vector_type(8)))  _Float16 h8;
typedef __attribute__((ext_vector_type(8)))  float    v8f;

static inline int ceildiv(int a, int b) { return (a + b - 1) / b; }

// ---- monotonic float<->uint encoding for atomicMax-based segment max ------
__device__ __forceinline__ unsigned encf(float f) {
  unsigned u = __float_as_uint(f);
  return (u & 0x80000000u) ? ~u : (u | 0x80000000u);
}
__device__ __forceinline__ float decf(unsigned u) {
  u = (u & 0x80000000u) ? (u & 0x7fffffffu) : ~u;
  return __uint_as_float(u);
}

__device__ __forceinline__ float geluf(float v) {
  return 0.5f * v * (1.f + erff(v * 0.70710678118f));
}

// ---- WMMA fragment loads (CDNA5 16-bit layouts, padded f16 operands) ------
// A 16x32: lane l (g=l>>4, m=l&15). VGPR j<4: K=kc+2j+g*8 ; j>=4: K=kc+16+2(j-4)+g*8
// => halves [kc+g*8 .. +7] then [kc+16+g*8 .. +7] : two contiguous 16B loads.
__device__ __forceinline__ v16h frag_a16(const _Float16* A, int lda, int m0,
                                         int kc, int lane) {
  const int g = lane >> 4;
  const _Float16* row = A + (size_t)(m0 + (lane & 15)) * lda + kc + g * 8;
  const h8 lo = *(const h8*)(row);
  const h8 hi = *(const h8*)(row + 16);
  v16h a;
#pragma unroll
  for (int i = 0; i < 8; ++i) { a[i] = lo[i]; a[8 + i] = hi[i]; }
  return a;
}

// B 32x16 (B = W^T, Wh row-major [Nout x ldw]): lane n=n0+(l&15), halves
// [kc+g*8 .. +7] then [kc+16+g*8 .. +7] : two contiguous 16B loads.
__device__ __forceinline__ v16h frag_b16(const _Float16* W, int ldw, int n0,
                                         int kc, int lane) {
  const int g = lane >> 4;
  const _Float16* row = W + (size_t)(n0 + (lane & 15)) * ldw + kc + g * 8;
  const h8 lo = *(const h8*)(row);
  const h8 hi = *(const h8*)(row + 16);
  v16h b;
#pragma unroll
  for (int i = 0; i < 8; ++i) { b[i] = lo[i]; b[8 + i] = hi[i]; }
  return b;
}

// ---- f32 -> zero-padded f16 repack (rows x K -> rows x Kp) ----------------
__global__ void cvt_pad_f16(const float* __restrict__ src, _Float16* __restrict__ dst,
                            int rows, int K, int Kp) {
  const int i = (int)(blockIdx.x * blockDim.x + threadIdx.x);
  if (i >= rows * Kp) return;
  const int r = i / Kp, k = i % Kp;
  dst[i] = (_Float16)((k < K) ? src[(size_t)r * K + k] : 0.f);
}

// ---------------------------------------------------------------------------
// Generic tiled GEMM:  C[M x Nout] = Ah[M x Kp] @ Wh[Nout x Kp]^T + bias (+C)
// Operands are padded f16. One wave per 16x16 output tile, M % 16 == 0.
// ---------------------------------------------------------------------------
template <bool ACC>
__global__ void gemm16_f16wmma(const _Float16* __restrict__ A, int Kp,
                               const _Float16* __restrict__ W,
                               const float* __restrict__ bias,
                               float* __restrict__ C, int M, int Nout) {
  const int lane   = threadIdx.x & 31;
  const int wave   = (int)((blockIdx.x * blockDim.x + threadIdx.x) >> 5);
  const int tilesN = Nout >> 4;
  const int tilesM = M >> 4;
  if (wave >= tilesM * tilesN) return;   // wave-uniform: EXEC stays all-ones
  const int m0 = (wave / tilesN) << 4;
  const int n0 = (wave % tilesN) << 4;

  v8f c = {};
  for (int kc = 0; kc < Kp; kc += 32) {
    v16h a = frag_a16(A, Kp, m0, kc, lane);
    v16h b = frag_b16(W, Kp, n0, kc, lane);
    c = __builtin_amdgcn_wmma_f32_16x16x32_f16(false, a, false, b, (short)0, c,
                                               false, false);
  }
  const int g = lane >> 4;
  const int n = n0 + (lane & 15);
  const float bv = bias ? bias[n] : 0.f;
#pragma unroll
  for (int r = 0; r < 8; ++r) {
    const int m = m0 + r + (g ? 8 : 0);
    const size_t idx = (size_t)m * Nout + n;
    float v = c[r] + bv;
    if (ACC) v += C[idx];
    C[idx] = v;
  }
}

// ---------------------------------------------------------------------------
// Fused per-edge MLP. One wave handles 16 edges end-to-end in LDS.
//  MODE 0: gravity  — build 15 LN'd features; 15->128 GELU ->64 GELU ->1 -> gs
//  MODE 1: edge_mlp — [h_s|h_t|ea] (138); ->128 ReLU ->64 ReLU ->1 -> res
//  MODE 2: gate     — [h_s|h_t|ea|gs] (139); ->64 ReLU ->1 -> sigmoid blend
// Weights W0/W1 are pre-padded f16 (ld = K0P / 128). 4 waves/block.
// ---------------------------------------------------------------------------
template <int MODE>
__global__ void edge_mlp_fused(const float* __restrict__ h,
                               const int* __restrict__ src,
                               const int* __restrict__ dst,
                               const float* __restrict__ ea,
                               const _Float16* __restrict__ W0, const float* __restrict__ b0,
                               const _Float16* __restrict__ W1, const float* __restrict__ b1,
                               const float* __restrict__ W2, const float* __restrict__ b2,
                               const float* __restrict__ ln_g, const float* __restrict__ ln_b,
                               const float* __restrict__ gs, const float* __restrict__ res,
                               float* __restrict__ out) {
  constexpr int K0  = (MODE == 0) ? 15 : (MODE == 1 ? 138 : 139);
  constexpr int K0P = (MODE == 0) ? 32 : 160;            // padded to 32 multiple
  constexpr int N0  = (MODE == 2) ? 64 : 128;            // first hidden width

  __shared__ alignas(16) _Float16 s_in[4][16 * K0P];
  __shared__ alignas(16) _Float16 s_a0[4][16 * N0];
  __shared__ alignas(16) _Float16 s_a1[4][16 * 64];

  const int lane = threadIdx.x & 31;
  const int w    = (int)(threadIdx.x >> 5);
  const int tile = (int)(blockIdx.x * (blockDim.x >> 5)) + w;
  if (tile >= N_EDGES / 16) return;
  const int e0 = tile * 16;

  // ---- stage input rows (f16) -------------------------------------------
  if constexpr (MODE == 0) {
    if (lane < 16) {
      const int e = e0 + lane;
      const int s = src[e], d = dst[e];
      float f[15];
#pragma unroll
      for (int i = 0; i < 4; ++i) {
        f[i]     = h[(size_t)s * 4 + i];
        f[4 + i] = h[(size_t)d * 4 + i];
      }
      f[8]  = f[0] + f[4];
      f[9]  = f[0] - f[4];
      f[10] = f[1] + f[5];
      f[11] = -ea[(size_t)e * 10 + 2];
      f[12] =  ea[(size_t)e * 10 + 3];
      f[13] =  ea[(size_t)e * 10 + 4];
      f[14] =  ea[(size_t)e * 10 + 5];
      float mu = 0.f;
#pragma unroll
      for (int i = 0; i < 15; ++i) mu += f[i];
      mu *= (1.f / 15.f);
      float var = 0.f;
#pragma unroll
      for (int i = 0; i < 15; ++i) { float t = f[i] - mu; var += t * t; }
      var *= (1.f / 15.f);
      const float rs = rsqrtf(var + 1e-5f);
      _Float16* row = &s_in[w][lane * K0P];
      for (int i = 0; i < K0P; ++i)
        row[i] = (_Float16)((i < 15) ? (ln_g[i] * (f[i] - mu) * rs + ln_b[i]) : 0.f);
    }
  } else {
    for (int ei = 0; ei < 16; ++ei) {
      const int e = e0 + ei;
      const int s = src[e], d = dst[e];
      for (int f = lane; f < K0P; f += 32) {
        float v = 0.f;
        if (f < 64)        v = h[(size_t)s * 64 + f];
        else if (f < 128)  v = h[(size_t)d * 64 + (f - 64)];
        else if (f < 138)  v = ea[(size_t)e * 10 + (f - 128)];
        else if (MODE == 2 && f == 138) v = gs[e];
        s_in[w][ei * K0P + f] = (_Float16)v;
      }
    }
  }

  // ---- layer 0: K0 -> N0, WMMA chain + activation -----------------------
#pragma unroll
  for (int tn = 0; tn < N0 / 16; ++tn) {
    v8f c = {};
    for (int kc = 0; kc < K0P; kc += 32) {
      v16h a = frag_a16(&s_in[w][0], K0P, 0, kc, lane);
      v16h b = frag_b16(W0, K0P, tn * 16, kc, lane);
      c = __builtin_amdgcn_wmma_f32_16x16x32_f16(false, a, false, b, (short)0, c,
                                                 false, false);
    }
    const int g = lane >> 4;
    const int n = tn * 16 + (lane & 15);
    const float bv = b0[n];
#pragma unroll
    for (int r = 0; r < 8; ++r) {
      const int m = r + (g ? 8 : 0);
      float v = c[r] + bv;
      v = (MODE == 0) ? geluf(v) : fmaxf(v, 0.f);
      s_a0[w][m * N0 + n] = (_Float16)v;
    }
  }

  // ---- layer 1: 128 -> 64 (MODE 0/1 only) -------------------------------
  if constexpr (MODE != 2) {
#pragma unroll
    for (int tn = 0; tn < 4; ++tn) {
      v8f c = {};
      for (int kc = 0; kc < N0; kc += 32) {
        v16h a = frag_a16(&s_a0[w][0], N0, 0, kc, lane);
        v16h b = frag_b16(W1, N0, tn * 16, kc, lane);
        c = __builtin_amdgcn_wmma_f32_16x16x32_f16(false, a, false, b, (short)0, c,
                                                   false, false);
      }
      const int g = lane >> 4;
      const int n = tn * 16 + (lane & 15);
      const float bv = b1[n];
#pragma unroll
      for (int r = 0; r < 8; ++r) {
        const int m = r + (g ? 8 : 0);
        float v = c[r] + bv;
        v = (MODE == 0) ? geluf(v) : fmaxf(v, 0.f);
        s_a1[w][m * 64 + n] = (_Float16)v;
      }
    }
  }

  // ---- final 64 -> 1 dot per edge ---------------------------------------
  if (lane < 16) {
    const int e = e0 + lane;
    const _Float16* row = (MODE == 2) ? &s_a0[w][lane * 64] : &s_a1[w][lane * 64];
    float acc = b2[0];
#pragma unroll
    for (int i = 0; i < 64; ++i) acc += (float)row[i] * W2[i];
    if constexpr (MODE == 2) {
      const float gt = 1.f / (1.f + expf(-acc));
      out[e] = gt * gs[e] + (1.f - gt) * res[e];
    } else {
      out[e] = acc;
    }
  }
}

// ---------------------------------------------------------------------------
// Attention (TransformerConv): warp per (edge, head), C = 64 channels.
// e_hc = ea . We[hd*64+c], recomputed on the fly (never materialized).
// ---------------------------------------------------------------------------
__global__ void attn_scores(const float* __restrict__ q, const float* __restrict__ k,
                            const float* __restrict__ We, const float* __restrict__ ea,
                            const int* __restrict__ src, const int* __restrict__ dst,
                            float* __restrict__ alpha, unsigned* __restrict__ menc,
                            int H) {
  const int wid  = (int)((blockIdx.x * blockDim.x + threadIdx.x) >> 5);
  const int lane = threadIdx.x & 31;
  if (wid >= N_EDGES * H) return;
  const int e = wid / H, hd = wid % H;
  const int s = src[e], d = dst[e];
  const int HC = H * HIDC;
  const float* qr = q + (size_t)d * HC + hd * HIDC;
  const float* kr = k + (size_t)s * HC + hd * HIDC;
  const float* wr = We + (size_t)hd * HIDC * 10;
  float eav[10];
#pragma unroll
  for (int f = 0; f < 10; ++f) eav[f] = ea[(size_t)e * 10 + f];
  float acc = 0.f;
#pragma unroll
  for (int t = 0; t < 2; ++t) {
    const int c = 2 * lane + t;
    float ec = 0.f;
#pragma unroll
    for (int f = 0; f < 10; ++f) ec += eav[f] * wr[c * 10 + f];
    acc += qr[c] * (kr[c] + ec);
  }
  for (int off = 16; off; off >>= 1) acc += __shfl_down(acc, off, 32);
  if (lane == 0) {
    const float a = acc * 0.125f;  // / sqrt(64)
    alpha[(size_t)e * H + hd] = a;
    atomicMax(&menc[(size_t)d * H + hd], encf(a));
  }
}

__global__ void attn_norm(const int* __restrict__ dst, float* __restrict__ alpha,
                          const unsigned* __restrict__ menc, float* __restrict__ z,
                          int H) {
  const int i = (int)(blockIdx.x * blockDim.x + threadIdx.x);
  if (i >= N_EDGES * H) return;
  const int e = i / H, hd = i % H;
  const int d = dst[e];
  const float ex = expf(alpha[i] - decf(menc[(size_t)d * H + hd]));
  alpha[i] = ex;
  atomicAdd(&z[(size_t)d * H + hd], ex);
}

__global__ void attn_aggregate(const float* __restrict__ v, const float* __restrict__ We,
                               const float* __restrict__ ea,
                               const int* __restrict__ src, const int* __restrict__ dst,
                               const float* __restrict__ alpha, const float* __restrict__ z,
                               float* __restrict__ hn, int H) {
  const int wid  = (int)((blockIdx.x * blockDim.x + threadIdx.x) >> 5);
  const int lane = threadIdx.x & 31;
  if (wid >= N_EDGES * H) return;
  const int e = wid / H, hd = wid % H;
  const int s = src[e], d = dst[e];
  const int HC = H * HIDC;
  const float wgt = alpha[(size_t)e * H + hd] / (z[(size_t)d * H + hd] + 1e-16f);
  const float* vr = v + (size_t)s * HC + hd * HIDC;
  const float* wr = We + (size_t)hd * HIDC * 10;
  float eav[10];
#pragma unroll
  for (int f = 0; f < 10; ++f) eav[f] = ea[(size_t)e * 10 + f];
#pragma unroll
  for (int t = 0; t < 2; ++t) {
    const int c = 2 * lane + t;
    float ec = 0.f;
#pragma unroll
    for (int f = 0; f < 10; ++f) ec += eav[f] * wr[c * 10 + f];
    atomicAdd(&hn[(size_t)d * HC + hd * HIDC + c], (vr[c] + ec) * wgt);
  }
}

// ---------------------------------------------------------------------------
// BatchNorm (training-mode batch statistics, biased var) + ELU
// ---------------------------------------------------------------------------
__global__ void bn_stats(const float* __restrict__ h, int D,
                         float* __restrict__ mean, float* __restrict__ rstd) {
  const int ch = blockIdx.x;
  __shared__ float s1[256], s2[256];
  float a = 0.f, b = 0.f;
  for (int i = threadIdx.x; i < N_NODES; i += blockDim.x) {
    const float v = h[(size_t)i * D + ch];
    a += v; b += v * v;
  }
  s1[threadIdx.x] = a; s2[threadIdx.x] = b;
  __syncthreads();
  for (int s = 128; s; s >>= 1) {
    if ((int)threadIdx.x < s) {
      s1[threadIdx.x] += s1[threadIdx.x + s];
      s2[threadIdx.x] += s2[threadIdx.x + s];
    }
    __syncthreads();
  }
  if (threadIdx.x == 0) {
    const float mu = s1[0] / (float)N_NODES;
    const float var = s2[0] / (float)N_NODES - mu * mu;
    mean[ch] = mu;
    rstd[ch] = rsqrtf(var + 1e-5f);
  }
}

__global__ void bn_apply_elu(float* __restrict__ h, const float* __restrict__ g,
                             const float* __restrict__ b, const float* __restrict__ mean,
                             const float* __restrict__ rstd, int D) {
  const int i = (int)(blockIdx.x * blockDim.x + threadIdx.x);
  if (i >= N_NODES * D) return;
  const int ch = i % D;
  const float v = (h[i] - mean[ch]) * rstd[ch] * g[ch] + b[ch];
  h[i] = v > 0.f ? v : expm1f(v);
}

__global__ void fill_f32(float* p, float v, int n) {
  const int i = (int)(blockIdx.x * blockDim.x + threadIdx.x);
  if (i < n) p[i] = v;
}
__global__ void fill_u32(unsigned* p, unsigned v, int n) {
  const int i = (int)(blockIdx.x * blockDim.x + threadIdx.x);
  if (i < n) p[i] = v;
}

// ---------------------------------------------------------------------------
// Host-side orchestration.
// Input indexing assumes jax pytree flattening (dict keys sorted):
//   0:x 1:edge_index 2:edge_attr
//   bns:   3..8   (bn{0,1,2}.{b,g})
//   convs: 9..35  (per conv: We, k.W, k.b, q.W, q.b, skip.W, skip.b, v.W, v.b)
//   edge_mlp: 36..41 (l0.W,l0.b,l1.W,l1.b,l2.W,l2.b)
//   gate:     42..45 (l0.W,l0.b,l1.W,l1.b)
//   grav:     46..53 (l0.W,l0.b,l1.W,l1.b,l2.W,l2.b,ln_b,ln_g)
// ---------------------------------------------------------------------------
extern "C" void kernel_launch(void* const* d_in, const int* in_sizes, int n_in,
                              void* d_out, int out_size, void* d_ws, size_t ws_size,
                              hipStream_t stream) {
  (void)in_sizes; (void)n_in; (void)out_size; (void)ws_size;
  const float* x  = (const float*)d_in[0];
  const int*   ei = (const int*)d_in[1];
  const float* ea = (const float*)d_in[2];
  const int* src = ei;
  const int* dst = ei + N_EDGES;

  const float* bn_b[3] = {(const float*)d_in[3], (const float*)d_in[5], (const float*)d_in[7]};
  const float* bn_g[3] = {(const float*)d_in[4], (const float*)d_in[6], (const float*)d_in[8]};

  const float *cWe[3], *ckW[3], *ckb[3], *cqW[3], *cqb[3], *csW[3], *csb[3], *cvW[3], *cvb[3];
  for (int i = 0; i < 3; ++i) {
    const int base = 9 + i * 9;
    cWe[i] = (const float*)d_in[base + 0];
    ckW[i] = (const float*)d_in[base + 1];  ckb[i] = (const float*)d_in[base + 2];
    cqW[i] = (const float*)d_in[base + 3];  cqb[i] = (const float*)d_in[base + 4];
    csW[i] = (const float*)d_in[base + 5];  csb[i] = (const float*)d_in[base + 6];
    cvW[i] = (const float*)d_in[base + 7];  cvb[i] = (const float*)d_in[base + 8];
  }
  const float* em_W0 = (const float*)d_in[36]; const float* em_b0 = (const float*)d_in[37];
  const float* em_W1 = (const float*)d_in[38]; const float* em_b1 = (const float*)d_in[39];
  const float* em_W2 = (const float*)d_in[40]; const float* em_b2 = (const float*)d_in[41];
  const float* gt_W0 = (const float*)d_in[42]; const float* gt_b0 = (const float*)d_in[43];
  const float* gt_W1 = (const float*)d_in[44]; const float* gt_b1 = (const float*)d_in[45];
  const float* gr_W0 = (const float*)d_in[46]; const float* gr_b0 = (const float*)d_in[47];
  const float* gr_W1 = (const float*)d_in[48]; const float* gr_b1 = (const float*)d_in[49];
  const float* gr_W2 = (const float*)d_in[50]; const float* gr_b2 = (const float*)d_in[51];
  const float* gr_lnb = (const float*)d_in[52];
  const float* gr_lng = (const float*)d_in[53];

  // ---- workspace carving: f32 region, then 16B-aligned f16 region --------
  float* ws = (float*)d_ws;
  float* gs    = ws; ws += N_EDGES;
  float* res   = ws; ws += N_EDGES;
  float* hA    = ws; ws += (size_t)N_NODES * 256;
  float* hB    = ws; ws += (size_t)N_NODES * 256;
  float* qbuf  = ws; ws += (size_t)N_NODES * 256;
  float* kbuf  = ws; ws += (size_t)N_NODES * 256;
  float* vbuf  = ws; ws += (size_t)N_NODES * 256;
  float* alpha = ws; ws += (size_t)N_EDGES * 4;
  float* zbuf  = ws; ws += (size_t)N_NODES * 4;
  unsigned* menc = (unsigned*)ws; ws += (size_t)N_NODES * 4;
  float* mean  = ws; ws += 256;
  float* rstd  = ws; ws += 256;

  _Float16* hws = (_Float16*)ws;                        // 4B-aligned, sizes %8==0
  _Float16* xh   = hws; hws += (size_t)N_NODES * 32;    // x padded 4->32
  _Float16* hh   = hws; hws += (size_t)N_NODES * 256;   // layer activations f16
  _Float16 *qWh[3], *kWh[3], *vWh[3], *sWh[3];
  const int dins[3]  = {4, 256, 256};
  const int heads[3] = {4, 4, 1};
  for (int L = 0; L < 3; ++L) {
    const int Kp = (L == 0) ? 32 : 256;
    const int dout = heads[L] * HIDC;
    qWh[L] = hws; hws += (size_t)dout * Kp;
    kWh[L] = hws; hws += (size_t)dout * Kp;
    vWh[L] = hws; hws += (size_t)dout * Kp;
    sWh[L] = hws; hws += (size_t)dout * Kp;
  }
  _Float16* grW0h = hws; hws += 128 * 32;
  _Float16* grW1h = hws; hws += 64 * 128;
  _Float16* emW0h = hws; hws += 128 * 160;
  _Float16* emW1h = hws; hws += 64 * 128;
  _Float16* gtW0h = hws; hws += 64 * 160;

  // ---- 0) repack all WMMA operands to padded f16 -------------------------
#define CVT(srcp, dstp, R, K, KP)                                             \
  cvt_pad_f16<<<ceildiv((R) * (KP), 256), 256, 0, stream>>>(srcp, dstp, R, K, KP)
  CVT(x, xh, N_NODES, 4, 32);
  for (int L = 0; L < 3; ++L) {
    const int Kp = (L == 0) ? 32 : 256;
    const int dout = heads[L] * HIDC;
    CVT(cqW[L], qWh[L], dout, dins[L], Kp);
    CVT(ckW[L], kWh[L], dout, dins[L], Kp);
    CVT(cvW[L], vWh[L], dout, dins[L], Kp);
    CVT(csW[L], sWh[L], dout, dins[L], Kp);
  }
  CVT(gr_W0, grW0h, 128, 15, 32);
  CVT(gr_W1, grW1h, 64, 128, 128);
  CVT(em_W0, emW0h, 128, 138, 160);
  CVT(em_W1, emW1h, 64, 128, 128);
  CVT(gt_W0, gtW0h, 64, 139, 160);
#undef CVT

  const int EDGE_TILES_BLOCKS = N_EDGES / 16 / 4;  // 3125 (4 waves/block)

  // ---- 1) gravity scores gs[e] ------------------------------------------
  edge_mlp_fused<0><<<EDGE_TILES_BLOCKS, 128, 0, stream>>>(
      x, src, dst, ea, grW0h, gr_b0, grW1h, gr_b1, gr_W2, gr_b2,
      gr_lng, gr_lnb, nullptr, nullptr, gs);

  // ---- 2) three TransformerConv layers ----------------------------------
  const float* hin = x;
  float* hout = hA;
  float* hother = hB;
  for (int L = 0; L < 3; ++L) {
    const int din = dins[L];
    const int Kp = (L == 0) ? 32 : 256;
    const int H = heads[L];
    const int dout = H * HIDC;
    const int waves = (N_NODES / 16) * (dout / 16);
    const int gblk = ceildiv(waves, 8);

    const _Float16* Ah;
    if (L == 0) {
      Ah = xh;
    } else {
      cvt_pad_f16<<<ceildiv(N_NODES * 256, 256), 256, 0, stream>>>(hin, hh,
                                                                   N_NODES, 256, 256);
      Ah = hh;
    }

    gemm16_f16wmma<false><<<gblk, 256, 0, stream>>>(Ah, Kp, qWh[L], cqb[L], qbuf, N_NODES, dout);
    gemm16_f16wmma<false><<<gblk, 256, 0, stream>>>(Ah, Kp, kWh[L], ckb[L], kbuf, N_NODES, dout);
    gemm16_f16wmma<false><<<gblk, 256, 0, stream>>>(Ah, Kp, vWh[L], cvb[L], vbuf, N_NODES, dout);

    fill_f32<<<ceildiv(N_NODES * dout, 256), 256, 0, stream>>>(hout, 0.f, N_NODES * dout);
    fill_u32<<<ceildiv(N_NODES * H, 256), 256, 0, stream>>>(menc, 0u, N_NODES * H);
    fill_f32<<<ceildiv(N_NODES * H, 256), 256, 0, stream>>>(zbuf, 0.f, N_NODES * H);

    const int eh = N_EDGES * H;
    attn_scores<<<ceildiv(eh, 8), 256, 0, stream>>>(qbuf, kbuf, cWe[L], ea, src, dst,
                                                    alpha, menc, H);
    attn_norm<<<ceildiv(eh, 256), 256, 0, stream>>>(dst, alpha, menc, zbuf, H);
    attn_aggregate<<<ceildiv(eh, 8), 256, 0, stream>>>(vbuf, cWe[L], ea, src, dst,
                                                       alpha, zbuf, hout, H);
    // skip connection accumulates into aggregated messages
    gemm16_f16wmma<true><<<gblk, 256, 0, stream>>>(Ah, Kp, sWh[L], csb[L], hout, N_NODES, dout);

    bn_stats<<<dout, 256, 0, stream>>>(hout, dout, mean, rstd);
    bn_apply_elu<<<ceildiv(N_NODES * dout, 256), 256, 0, stream>>>(hout, bn_g[L], bn_b[L],
                                                                   mean, rstd, dout);
    hin = hout;
    float* t = hout; hout = hother; hother = t;
    (void)din;
  }
  const float* hfin = hin;  // [N x 64] after layer 2

  // ---- 3) residual edge MLP and gated blend -> d_out ---------------------
  edge_mlp_fused<1><<<EDGE_TILES_BLOCKS, 128, 0, stream>>>(
      hfin, src, dst, ea, emW0h, em_b0, emW1h, em_b1, em_W2, em_b2,
      nullptr, nullptr, nullptr, nullptr, res);
  edge_mlp_fused<2><<<EDGE_TILES_BLOCKS, 128, 0, stream>>>(
      hfin, src, dst, ea, gtW0h, gt_b0, nullptr, nullptr, gt_W1, gt_b1,
      nullptr, nullptr, gs, res, (float*)d_out);
}
